// MixedTransformer_87222195847594
// MI455X (gfx1250) — matrix operand order repaired and checked by reading, not compiled, not run on hardware
//
#include <hip/hip_runtime.h>

#define BS     2
#define ERA    35718
#define HMESH  10242
#define NE2H   107154
#define NH2H   61440
#define NH2E   107154
#define MG     (BS*ERA)     // 71436 grid nodes total
#define MH     (BS*HMESH)   // 20484 mesh nodes total
#define MGPAD  71440
#define MHPAD  20496

typedef __attribute__((ext_vector_type(16))) _Float16 v16h;
typedef __attribute__((ext_vector_type(8)))  _Float16 v8h;
typedef __attribute__((ext_vector_type(8)))  float    v8f;

static __device__ __forceinline__ unsigned f2key(float f) {
  unsigned u = __float_as_uint(f);
  return (u & 0x80000000u) ? ~u : (u | 0x80000000u);
}
static __device__ __forceinline__ float key2f(unsigned k) {
  unsigned u = (k & 0x80000000u) ? (k ^ 0x80000000u) : ~k;
  return __uint_as_float(u);
}
static __device__ __forceinline__ float leaky(float x) { return x > 0.f ? x : 0.2f * x; }

// ---------------- generic fill ----------------
__global__ void k_fill(unsigned* p, unsigned v, int n) {
  int t = blockIdx.x * blockDim.x + threadIdx.x;
  if (t < n) p[t] = v;
}

// ---------------- small GEMV fold: out[r] = sum_c W[r*rs + c*cs] * v[c] ----------------
__global__ void k_fold(const float* __restrict__ W, const float* __restrict__ v,
                       float* __restrict__ out, int R, int C, int rs, int cs) {
  int r = blockIdx.x * blockDim.x + threadIdx.x;
  if (r >= R) return;
  float s = 0.f;
  for (int c = 0; c < C; ++c) s += W[(size_t)r * rs + (size_t)c * cs] * v[c];
  out[r] = s;
}

// ---------------- weight f32(KxN) -> transposed padded f16(N x Kpad) ----------------
__global__ void k_wT16(const float* __restrict__ W, _Float16* __restrict__ Bt,
                       int K, int N, int Kpad) {
  int t = blockIdx.x * blockDim.x + threadIdx.x;
  if (t >= N * Kpad) return;
  int n = t / Kpad, k = t - n * Kpad;
  Bt[t] = (k < K) ? (_Float16)W[(size_t)k * N + n] : (_Float16)0.f;
}

// ---------------- WMMA GEMM: C(MxN) = A(Mpad x Kpad f16) * Bt(N x Kpad f16)^T ----------------
__global__ void k_gemm_wmma(const _Float16* __restrict__ A, const _Float16* __restrict__ Bt,
                            float* __restrict__ C, int M, int N, int Kpad) {
  int tiles_n = N >> 4;
  int tile = blockIdx.x;
  int tm = tile / tiles_n, tn = tile - tm * tiles_n;
  int lane = threadIdx.x & 31;
  int hi = lane >> 4;        // lane group 0/1
  int lo = lane & 15;
  const _Float16* arow = A  + (size_t)(tm * 16 + lo) * Kpad;
  const _Float16* brow = Bt + (size_t)(tn * 16 + lo) * Kpad;
  v8f acc = {};
  for (int kb = 0; kb < Kpad; kb += 32) {
    __builtin_prefetch(arow + kb + 64, 0, 0);
    // A fragment (16x32 f16): lane row = lo; halves: K = kb + (i/8)*16 + hi*8 + i%8
    union { v16h v; v8h h[2]; } a;
    a.h[0] = *(const v8h*)(arow + kb + hi * 8);
    a.h[1] = *(const v8h*)(arow + kb + 16 + hi * 8);
    // B fragment (32x16 f16): lane col = lo; halves: K = kb + (i/8)*16 + hi*8 + i%8
    union { v16h v; v8h h[2]; } bb;
    bb.h[0] = *(const v8h*)(brow + kb + hi * 8);
    bb.h[1] = *(const v8h*)(brow + kb + 16 + hi * 8);
    v16h b = bb.v;
    acc = __builtin_amdgcn_wmma_f32_16x16x32_f16(false, a.v, false, b,
                                                 (short)0, acc, false, false);
  }
  int col = tn * 16 + lo;
  #pragma unroll
  for (int r = 0; r < 8; ++r) {
    int m = tm * 16 + hi * 8 + r;
    if (m < M) C[(size_t)m * N + col] = acc[r];
  }
}

// ---------------- encoder node prep: build A (f16, Kpad=128) + per-node logit ls ----------------
__global__ void k_enc_prep(const float* __restrict__ x, const float* __restrict__ latlon,
                           const float* __restrict__ av, _Float16* __restrict__ Ah,
                           float* __restrict__ ls) {
  int m = blockIdx.x * blockDim.x + threadIdx.x;
  if (m >= MGPAD) return;
  _Float16* a = Ah + (size_t)m * 128;
  if (m >= MG) { for (int k = 0; k < 128; ++k) a[k] = (_Float16)0.f; return; }
  int n = m % ERA;
  const float* xr = x + (size_t)m * 98;
  float acc = 0.f;
  for (int j = 0; j < 98; ++j) { float f = xr[j]; a[j] = (_Float16)f; acc += f * av[j]; }
  const float* ll = latlon + (size_t)n * 4;
  for (int j = 0; j < 4; ++j) { float f = ll[j]; a[98 + j] = (_Float16)f; acc += f * av[98 + j]; }
  for (int k = 102; k < 128; ++k) a[k] = (_Float16)0.f;
  ls[m] = acc;
}

// ---------------- ctx logit: lc[m] = ctx[n]*cv[0] + latlon[n]·cv[1..4] ----------------
__global__ void k_ctx(const float* __restrict__ ctx, const float* __restrict__ latlon,
                      const float* __restrict__ cv, float* __restrict__ lc,
                      int Mtot, int nper) {
  int m = blockIdx.x * blockDim.x + threadIdx.x;
  if (m >= Mtot) return;
  int n = m % nper;
  float s = ctx[n] * cv[0];
  for (int j = 0; j < 4; ++j) s += latlon[(size_t)n * 4 + j] * cv[1 + j];
  lc[m] = s;
}

// ---------------- mapper pass1: atomic max of logits per dst ----------------
__global__ void k_map_max(const int* __restrict__ idx, const float* __restrict__ attr,
                          const float* __restrict__ ev, const float* __restrict__ ls,
                          const float* __restrict__ lc, unsigned* __restrict__ mk,
                          int E, int sInc, int dInc, int nInst) {
  int ei = blockIdx.x * blockDim.x + threadIdx.x;
  if (ei >= nInst) return;
  int b = ei / E, e = ei - b * E;
  int s = idx[e] + b * sInc;
  int d = idx[E + e] + b * dInc;
  float le = attr[e*3]*ev[0] + attr[e*3+1]*ev[1] + attr[e*3+2]*ev[2];
  float lg = leaky(ls[s] + lc[d] + le);
  atomicMax(mk + d, f2key(lg));
}

// ---------------- mapper pass2: one wave per edge, scatter exp-weighted values ----------------
__global__ void k_map_scatter(const int* __restrict__ idx, const float* __restrict__ attr,
                              const float* __restrict__ ev, const float* __restrict__ ls,
                              const float* __restrict__ lc, const float* __restrict__ val,
                              const unsigned* __restrict__ mk, float* __restrict__ num,
                              float* __restrict__ den, int E, int sInc, int dInc,
                              int NC, int nInst) {
  int gt = blockIdx.x * blockDim.x + threadIdx.x;
  int ei = gt >> 5, lane = gt & 31;
  if (ei >= nInst) return;
  int b = ei / E, e = ei - b * E;
  int s = idx[e] + b * sInc;
  int d = idx[E + e] + b * dInc;
  float le = attr[e*3]*ev[0] + attr[e*3+1]*ev[1] + attr[e*3+2]*ev[2];
  float lg = leaky(ls[s] + lc[d] + le);
  float w = __expf(lg - key2f(mk[d]));
  if (lane == 0) atomicAdd(den + d, w);
  const float* vr = val + (size_t)s * NC;
  float* nr = num + (size_t)d * NC;
  for (int c = lane; c < NC; c += 32) atomicAdd(nr + c, w * vr[c]);
}

// ---------------- mapper finalize: out = num / (den + 1e-9) ----------------
__global__ void k_map_fin(const float* __restrict__ num, const float* __restrict__ den,
                          float* __restrict__ out, int NC, int total) {
  int t = blockIdx.x * blockDim.x + threadIdx.x;
  if (t >= total) return;
  out[t] = num[t] / (den[t / NC] + 1e-9f);
}

// ---------------- H f32 -> f16 (optionally gelu), zero-pad rows ----------------
__global__ void k_h16(const float* __restrict__ H, _Float16* __restrict__ Hh,
                      int Mtot, int Mpad, int gelu) {
  int t = blockIdx.x * blockDim.x + threadIdx.x;
  if (t >= Mpad * 256) return;
  int m = t >> 8;
  float v = (m < Mtot) ? H[t] : 0.f;
  if (gelu) v = 0.5f * v * (1.f + tanhf(0.7978845608f * (v + 0.044715f * v * v * v)));
  Hh[t] = (_Float16)v;
}

// ---------------- GAT per-node attention dots: qs/qd (node, head) ----------------
__global__ void k_gat_dots(const float* __restrict__ q, const float* __restrict__ as,
                           const float* __restrict__ ad, float* __restrict__ qs,
                           float* __restrict__ qd) {
  int t = blockIdx.x * blockDim.x + threadIdx.x;
  if (t >= MH * 2) return;
  int n = t >> 1, h = t & 1;
  const float* qr = q + (size_t)n * 256 + h * 128;
  float ss = 0.f, sd = 0.f;
  for (int i = 0; i < 128; ++i) { float v = qr[i]; ss += v * as[h*128+i]; sd += v * ad[h*128+i]; }
  qs[t] = ss; qd[t] = sd;
}

// ---------------- GAT pass1 (per edge-instance per head) ----------------
__global__ void k_gat_max(const int* __restrict__ idx, const float* __restrict__ attr,
                          const float* __restrict__ gef, const float* __restrict__ qs,
                          const float* __restrict__ qd, unsigned* __restrict__ mk,
                          int E, int inc, int nInst2) {
  int t = blockIdx.x * blockDim.x + threadIdx.x;
  if (t >= nInst2) return;
  int h = t & 1, ei = t >> 1;
  int b = ei / E, e = ei - b * E;
  int s = idx[e] + b * inc;
  int d = idx[E + e] + b * inc;
  float le = attr[e*3]*gef[h*3] + attr[e*3+1]*gef[h*3+1] + attr[e*3+2]*gef[h*3+2];
  float lg = leaky(qs[s*2+h] + qd[d*2+h] + le);
  atomicMax(mk + d * 2 + h, f2key(lg));
}

// ---------------- GAT pass2: one wave per edge-instance ----------------
__global__ void k_gat_scatter(const int* __restrict__ idx, const float* __restrict__ attr,
                              const float* __restrict__ gef, const float* __restrict__ qs,
                              const float* __restrict__ qd, const float* __restrict__ q,
                              const unsigned* __restrict__ mk, float* __restrict__ num,
                              float* __restrict__ den, int E, int inc, int nInst) {
  int gt = blockIdx.x * blockDim.x + threadIdx.x;
  int ei = gt >> 5, lane = gt & 31;
  if (ei >= nInst) return;
  int b = ei / E, e = ei - b * E;
  int s = idx[e] + b * inc;
  int d = idx[E + e] + b * inc;
  float w[2];
  #pragma unroll
  for (int h = 0; h < 2; ++h) {
    float le = attr[e*3]*gef[h*3] + attr[e*3+1]*gef[h*3+1] + attr[e*3+2]*gef[h*3+2];
    float lg = leaky(qs[s*2+h] + qd[d*2+h] + le);
    w[h] = __expf(lg - key2f(mk[d*2+h]));
  }
  if (lane < 2) atomicAdd(den + d * 2 + lane, w[lane]);
  const float* qr = q + (size_t)s * 256;
  float* nr = num + (size_t)d * 256;
  for (int c = lane; c < 256; c += 32) atomicAdd(nr + c, w[c >> 7] * qr[c]);
}

__global__ void k_gat_fin(const float* __restrict__ num, const float* __restrict__ den,
                          float* __restrict__ out) {
  int t = blockIdx.x * blockDim.x + threadIdx.x;
  if (t >= MH * 256) return;
  out[t] = num[t] / (den[(t >> 8) * 2 + ((t & 255) >> 7)] + 1e-9f);
}

// ---------------- decoder prep: xp = h+x_lat | latlon -> f16 Kpad=288, ls_dec ----------------
__global__ void k_dec_prep(const float* __restrict__ h1, const float* __restrict__ xlat,
                           const float* __restrict__ latlon, const float* __restrict__ av,
                           _Float16* __restrict__ Ah, float* __restrict__ ls) {
  int m = blockIdx.x * blockDim.x + threadIdx.x;
  if (m >= MHPAD) return;
  _Float16* a = Ah + (size_t)m * 288;
  if (m >= MH) { for (int k = 0; k < 288; ++k) a[k] = (_Float16)0.f; return; }
  int n = m % HMESH;
  float acc = 0.f;
  const float* hr = h1 + (size_t)m * 256;
  const float* xr = xlat + (size_t)m * 256;
  for (int j = 0; j < 256; ++j) { float f = hr[j] + xr[j]; a[j] = (_Float16)f; acc += f * av[j]; }
  const float* ll = latlon + (size_t)n * 4;
  for (int j = 0; j < 4; ++j) { float f = ll[j]; a[256 + j] = (_Float16)f; acc += f * av[256 + j]; }
  for (int k = 260; k < 288; ++k) a[k] = (_Float16)0.f;
  ls[m] = acc;
}

// ---------------- final output: out = num/(den+1e-9) + x[..., :96] ----------------
__global__ void k_final(const float* __restrict__ num, const float* __restrict__ den,
                        const float* __restrict__ x, float* __restrict__ out) {
  int t = blockIdx.x * blockDim.x + threadIdx.x;
  if (t >= MG * 96) return;
  int m = t / 96, c = t - m * 96;
  out[t] = num[t] / (den[m] + 1e-9f) + x[(size_t)m * 98 + c];
}

extern "C" void kernel_launch(void* const* d_in, const int* in_sizes, int n_in,
                              void* d_out, int out_size, void* d_ws, size_t ws_size,
                              hipStream_t stream) {
  const float* x        = (const float*)d_in[0];
  const int*   e2h_idx  = (const int*)d_in[1];
  const int*   h2h_idx  = (const int*)d_in[2];
  const int*   h2e_idx  = (const int*)d_in[3];
  const float* e2h_attr = (const float*)d_in[4];
  const float* h2h_attr = (const float*)d_in[5];
  const float* h2e_attr = (const float*)d_in[6];
  const float* era_ll   = (const float*)d_in[7];
  const float* h_ll     = (const float*)d_in[8];
  const float* fm_ctx   = (const float*)d_in[9];
  const float* fm_Wsrc  = (const float*)d_in[10];
  const float* fm_Wctx  = (const float*)d_in[11];
  const float* fm_Wedge = (const float*)d_in[12];
  const float* fm_att   = (const float*)d_in[13];
  const float* fm_Wval  = (const float*)d_in[14];
  const float* bm_ctx   = (const float*)d_in[15];
  const float* bm_Wsrc  = (const float*)d_in[16];
  const float* bm_Wctx  = (const float*)d_in[17];
  const float* bm_Wedge = (const float*)d_in[18];
  const float* bm_att   = (const float*)d_in[19];
  const float* bm_Wval  = (const float*)d_in[20];
  const float* gat_W    = (const float*)d_in[21];
  const float* gat_We   = (const float*)d_in[22];
  const float* gat_asrc = (const float*)d_in[23];
  const float* gat_adst = (const float*)d_in[24];
  float* out = (float*)d_out;

  // ---- workspace carve-out ----
  size_t off = 0;
  auto alloc = [&](size_t bytes) -> char* {
    char* p = (char*)d_ws + off;
    off += (bytes + 255) & ~(size_t)255;
    return p;
  };
  float*     folds    = (float*)alloc(1024 * 4);          // packed small fold vectors
  _Float16*  WvT_enc  = (_Float16*)alloc((size_t)256 * 128 * 2);
  _Float16*  WvT_dec  = (_Float16*)alloc((size_t)96 * 288 * 2);
  _Float16*  gWT      = (_Float16*)alloc((size_t)2 * 256 * 256 * 2);
  _Float16*  encAh    = (_Float16*)alloc((size_t)MGPAD * 128 * 2);
  float*     ls_enc   = (float*)alloc((size_t)MG * 4);
  float*     lc_enc   = (float*)alloc((size_t)MH * 4);
  float*     val_enc  = (float*)alloc((size_t)MG * 256 * 4);
  unsigned*  mk       = (unsigned*)alloc((size_t)MG * 4);  // shared max-keys (max need = MG)
  float*     den      = (float*)alloc((size_t)MG * 4);     // shared denominators
  float*     num      = (float*)alloc((size_t)MG * 96 * 4 > (size_t)MH * 256 * 4
                                      ? (size_t)MG * 96 * 4 : (size_t)MH * 256 * 4);
  float*     x_lat    = (float*)alloc((size_t)MH * 256 * 4);
  _Float16*  Hh       = (_Float16*)alloc((size_t)MHPAD * 256 * 2);
  float*     q        = (float*)alloc((size_t)MH * 256 * 4);
  float*     qs       = (float*)alloc((size_t)MH * 2 * 4);
  float*     qd       = (float*)alloc((size_t)MH * 2 * 4);
  float*     hout     = (float*)alloc((size_t)MH * 256 * 4);
  _Float16*  decAh    = (_Float16*)alloc((size_t)MHPAD * 288 * 2);
  float*     val_dec  = (float*)alloc((size_t)MH * 96 * 4);
  float*     ls_dec   = (float*)alloc((size_t)MH * 4);
  float*     lc_dec   = (float*)alloc((size_t)MG * 4);
  (void)ws_size; (void)n_in; (void)in_sizes; (void)out_size;

  // fold vector layout inside `folds`
  float* fm_av = folds + 0;    // 102
  float* fm_cv = folds + 128;  // 5
  float* fm_ev = folds + 136;  // 3
  float* bm_av = folds + 160;  // 260
  float* bm_cv = folds + 432;  // 5
  float* bm_ev = folds + 440;  // 3
  float* gef   = folds + 448;  // L*H*3 = 12

  auto cdiv = [](long a, long b) { return (unsigned)((a + b - 1) / b); };

  // ---- stage 0: fold vectors + f16 transposed weights ----
  k_fold<<<1, 128, 0, stream>>>(fm_Wsrc, fm_att, fm_av, 102, 256, 256, 1);
  k_fold<<<1, 32, 0, stream>>>(fm_Wctx, fm_att, fm_cv, 5, 256, 256, 1);
  k_fold<<<1, 32, 0, stream>>>(fm_Wedge, fm_att, fm_ev, 3, 256, 256, 1);
  k_fold<<<2, 160, 0, stream>>>(bm_Wsrc, bm_att, bm_av, 260, 96, 96, 1);
  k_fold<<<1, 32, 0, stream>>>(bm_Wctx, bm_att, bm_cv, 5, 96, 96, 1);
  k_fold<<<1, 32, 0, stream>>>(bm_Wedge, bm_att, bm_ev, 3, 96, 96, 1);
  for (int l = 0; l < 2; ++l)
    for (int h = 0; h < 2; ++h) {
      const float* aeg = (const float*)d_in[25] + l * 256 + h * 128;  // gat_aedge
      k_fold<<<1, 32, 0, stream>>>(gat_We + l * 768 + h * 128, aeg,
                                   gef + (l * 2 + h) * 3, 3, 128, 256, 1);
    }
  k_wT16<<<cdiv(256 * 128, 256), 256, 0, stream>>>(fm_Wval, WvT_enc, 102, 256, 128);
  k_wT16<<<cdiv(96 * 288, 256), 256, 0, stream>>>(bm_Wval, WvT_dec, 260, 96, 288);
  for (int l = 0; l < 2; ++l)
    k_wT16<<<cdiv(256 * 256, 256), 256, 0, stream>>>(gat_W + (size_t)l * 65536,
                                                     gWT + (size_t)l * 65536, 256, 256, 256);

  // ---- stage 1: encoder mapper (grid -> mesh) ----
  k_enc_prep<<<cdiv(MGPAD, 128), 128, 0, stream>>>(x, era_ll, fm_av, encAh, ls_enc);
  k_ctx<<<cdiv(MH, 256), 256, 0, stream>>>(fm_ctx, h_ll, fm_cv, lc_enc, MH, HMESH);
  k_gemm_wmma<<<(MGPAD / 16) * (256 / 16), 32, 0, stream>>>(encAh, WvT_enc, val_enc, MG, 256, 128);
  k_fill<<<cdiv(MH, 256), 256, 0, stream>>>(mk, 0u, MH);
  k_fill<<<cdiv(MH, 256), 256, 0, stream>>>((unsigned*)den, 0u, MH);
  k_fill<<<cdiv(MH * 256, 256), 256, 0, stream>>>((unsigned*)num, 0u, MH * 256);
  k_map_max<<<cdiv(BS * NE2H, 256), 256, 0, stream>>>(e2h_idx, e2h_attr, fm_ev, ls_enc, lc_enc,
                                                      mk, NE2H, ERA, HMESH, BS * NE2H);
  k_map_scatter<<<cdiv((long)BS * NE2H * 32, 256), 256, 0, stream>>>(
      e2h_idx, e2h_attr, fm_ev, ls_enc, lc_enc, val_enc, mk, num, den,
      NE2H, ERA, HMESH, 256, BS * NE2H);
  k_map_fin<<<cdiv(MH * 256, 256), 256, 0, stream>>>(num, den, x_lat, 256, MH * 256);

  // ---- stage 2: GAT processor (2 layers, last has no gelu; residual added in decoder prep) ----
  for (int l = 0; l < 2; ++l) {
    k_h16<<<cdiv(MHPAD * 256, 256), 256, 0, stream>>>(l == 0 ? x_lat : hout, Hh, MH, MHPAD,
                                                      l == 0 ? 0 : 1);
    k_gemm_wmma<<<(MHPAD / 16) * (256 / 16), 32, 0, stream>>>(Hh, gWT + (size_t)l * 65536,
                                                              q, MH, 256, 256);
    k_gat_dots<<<cdiv(MH * 2, 256), 256, 0, stream>>>(q, gat_asrc + l * 256,
                                                      gat_adst + l * 256, qs, qd);
    k_fill<<<cdiv(MH * 2, 256), 256, 0, stream>>>(mk, 0u, MH * 2);
    k_fill<<<cdiv(MH * 2, 256), 256, 0, stream>>>((unsigned*)den, 0u, MH * 2);
    k_fill<<<cdiv(MH * 256, 256), 256, 0, stream>>>((unsigned*)num, 0u, MH * 256);
    k_gat_max<<<cdiv((long)BS * NH2H * 2, 256), 256, 0, stream>>>(
        h2h_idx, h2h_attr, gef + l * 6, qs, qd, mk, NH2H, HMESH, BS * NH2H * 2);
    k_gat_scatter<<<cdiv((long)BS * NH2H * 32, 256), 256, 0, stream>>>(
        h2h_idx, h2h_attr, gef + l * 6, qs, qd, q, mk, num, den, NH2H, HMESH, BS * NH2H);
    k_gat_fin<<<cdiv(MH * 256, 256), 256, 0, stream>>>(num, den, hout);
  }

  // ---- stage 3: decoder mapper (mesh -> grid) + residual output ----
  k_dec_prep<<<cdiv(MHPAD, 128), 128, 0, stream>>>(hout, x_lat, h_ll, bm_av, decAh, ls_dec);
  k_ctx<<<cdiv(MG, 256), 256, 0, stream>>>(bm_ctx, era_ll, bm_cv, lc_dec, MG, ERA);
  k_gemm_wmma<<<(MHPAD / 16) * (96 / 16), 32, 0, stream>>>(decAh, WvT_dec, val_dec, MH, 96, 288);
  k_fill<<<cdiv(MG, 256), 256, 0, stream>>>(mk, 0u, MG);
  k_fill<<<cdiv(MG, 256), 256, 0, stream>>>((unsigned*)den, 0u, MG);
  k_fill<<<cdiv(MG * 96, 256), 256, 0, stream>>>((unsigned*)num, 0u, MG * 96);
  k_map_max<<<cdiv(BS * NH2E, 256), 256, 0, stream>>>(h2e_idx, h2e_attr, bm_ev, ls_dec, lc_dec,
                                                      mk, NH2E, HMESH, ERA, BS * NH2E);
  k_map_scatter<<<cdiv((long)BS * NH2E * 32, 256), 256, 0, stream>>>(
      h2e_idx, h2e_attr, bm_ev, ls_dec, lc_dec, val_dec, mk, num, den,
      NH2E, HMESH, ERA, 96, BS * NH2E);
  k_final<<<cdiv(MG * 96, 256), 256, 0, stream>>>(num, den, x, out);
}
